// LSTM_Text_Encoder_20675972563327
// MI455X (gfx1250) — matrix operand order repaired and checked by reading, not compiled, not run on hardware
//
#include <hip/hip_runtime.h>

// ---------------------------------------------------------------------------
// Bidirectional GRU text encoder for MI455X (gfx1250, wave32, WMMA bf16, TDM).
// T=128, B=512, E=1024, W=1024.
// Per step: G(512x3072) = [x_t | h] @ [Wi.T ; Wh.T] + (bi+bh)   (K=2048, bf16 WMMA)
//           Ghn(512x1024) = h @ Wh_n.T + bh_n                   (K=1024)
// Then pointwise GRU: r=sig(G_r), z=sig(G_z), n=tanh(G_n+(r-1)*Ghn),
//                     h' = (1-z)*n + z*h   (+ mask hold for backward dir)
// ---------------------------------------------------------------------------

typedef __attribute__((ext_vector_type(16))) __bf16   bf16x16;
typedef __attribute__((ext_vector_type(8)))  float    floatx8;
typedef __attribute__((ext_vector_type(4)))  unsigned uint32x4;
typedef __attribute__((ext_vector_type(4)))  float    floatx4;
typedef __attribute__((ext_vector_type(8)))  int      int32x8;
typedef __attribute__((ext_vector_type(4)))  int      int32x4;

static constexpr int TT = 128;
static constexpr int BB = 512;
static constexpr int WW = 1024;
static constexpr int EE = 1024;

union FragBF { uint32x4 u[2]; bf16x16 v; };

#if defined(__AMDGCN__) && __has_builtin(__builtin_amdgcn_tensor_load_to_lds)
#define USE_TDM 1
#else
#define USE_TDM 0
#endif

__device__ __forceinline__ unsigned short f2bf(float f) {
    unsigned int u = __float_as_uint(f);
    u += 0x7FFFu + ((u >> 16) & 1u);   // round-to-nearest-even
    return (unsigned short)(u >> 16);
}
__device__ __forceinline__ unsigned int pack_bf2(float a, float b) {
    return (unsigned)f2bf(a) | ((unsigned)f2bf(b) << 16);
}

__device__ __forceinline__ void wait_tensor0() {
#if defined(__AMDGCN__) && __has_builtin(__builtin_amdgcn_s_wait_tensorcnt)
    __builtin_amdgcn_s_wait_tensorcnt(0);
#elif defined(__AMDGCN__)
    asm volatile("s_wait_tensorcnt 0x0" ::: "memory");
#endif
}

#if USE_TDM
// Issue a TDM 2-D tile load: 128 rows x 32 bf16, row stride = strideElems,
// into LDS at ldsOff with 16B padding after every 64B row (matches [128][40]).
// Descriptor per CDNA5 ISA sect.8 (D# groups).
// This toolchain exposes the 6-arg builtin:
//   (uint32x4 g0, int32x8 g1, int32x4 g2, int32x4 g3, int32x8 g4, i32 cpol)
__device__ __forceinline__ void tdm_load_tile(unsigned ldsOff,
                                              const unsigned short* gsrc,
                                              unsigned strideElems)
{
    unsigned long long ga = (unsigned long long)gsrc;
    uint32x4 g0;
    g0.x = 1u;                                        // count=1 (valid), user mode, no gather
    g0.y = ldsOff;                                    // lds_addr[31:0]
    g0.z = (unsigned)(ga & 0xFFFFFFFFull);            // global_addr[31:0]
    g0.w = (unsigned)((ga >> 32) & 0x1FFFFFFull)      // global_addr[56:32]
         | (2u << 30);                                // type=2 ("image")
    int32x8 g1;
    g1[0] = (int)((1u << 16)                          // data_size = 2 bytes
                | (1u << 20)                          // pad_enable
                | (3u << 22)                          // pad_interval: 16 DWORDs (64B row)
                | (3u << 25));                        // pad_amount: 4 DWORDs (16B)
    g1[1] = 0;                                        // atomic_barrier=0, tensor_dim0[15:0]=0
    g1[2] = (int)0x10u;                               // tensor_dim0[31:16]=0x10 (1M), dim1 lo=0
    g1[3] = (int)((32u << 16) | 0x10u);               // tensor_dim1 hi=0x10 (1M), tile_dim0=32
    g1[4] = 128;                                      // tile_dim1=128, tile_dim2=0
    g1[5] = (int)strideElems;                         // tensor_dim0_stride[31:0]
    g1[6] = 0;                                        // stride hi / dim1_stride lo
    g1[7] = 0;                                        // dim1_stride hi
    const int32x4 gz4 = {0, 0, 0, 0};
    const int32x8 gz8 = {0, 0, 0, 0, 0, 0, 0, 0};
    __builtin_amdgcn_tensor_load_to_lds(g0, g1, gz4, gz4, gz8, 0);
}
#endif

// ---- fragment loads + 8 WMMAs for one 32-deep K-step (wave tile 64x32) ----
__device__ __forceinline__ void mma_tiles(const unsigned short (*As)[40],
                                          const unsigned short (*Bs)[40],
                                          int lane, int wm, int wn,
                                          floatx8 acc[4][2])
{
    FragBF a[4], b2[2];
    const int mrow0 = lane & 15;
    const int lo = (lane < 16) ? 0 : 8;      // A: lanes 0-15 K{0-7,16-23}, 16-31 K{8-15,24-31}
    #pragma unroll
    for (int mi = 0; mi < 4; ++mi) {
        const int m = wm + mi * 16 + mrow0;
        a[mi].u[0] = *(const uint32x4*)&As[m][lo];
        a[mi].u[1] = *(const uint32x4*)&As[m][lo + 16];
    }
    const int ko = (lane < 16) ? 0 : 8;      // B: lanes 0-15 K{0-7,16-23}, 16-31 K{8-15,24-31}
    #pragma unroll
    for (int ni = 0; ni < 2; ++ni) {
        const int n = wn + ni * 16 + mrow0;
        b2[ni].u[0] = *(const uint32x4*)&Bs[n][ko];
        b2[ni].u[1] = *(const uint32x4*)&Bs[n][ko + 16];
    }
    #pragma unroll
    for (int mi = 0; mi < 4; ++mi)
        #pragma unroll
        for (int ni = 0; ni < 2; ++ni)
            acc[mi][ni] = __builtin_amdgcn_wmma_f32_16x16x32_bf16(
                false, a[mi].v, false, b2[ni].v, (short)0, acc[mi][ni], false, false);
}

// ---------------------------------------------------------------------------
// Tiled bf16 WMMA GEMM, C[b][n] = sum_k A[b][k]*Bmat[n][k] + bias[n].
// Phase 1 (k < Ksplit): A rows gathered from embed (f32 -> bf16 on the fly).
// Phase 2 (k >= Ksplit): A rows from bf16 hidden state (TDM tile load).
// B tiles always via TDM tile load (fallback: vector loads + ds_store).
// blockIdx.z selects direction. Block: 256 threads = 8 waves; block tile
// 128x128; wave tile 64x32 (4x2 v_wmma_f32_16x16x32_bf16).
// ---------------------------------------------------------------------------
__global__ __launch_bounds__(256)
void gemm_step(const unsigned short* __restrict__ Bmat, long long BmatStride, int ldb,
               const float* __restrict__ bias0, const float* __restrict__ bias1,
               float* __restrict__ Cout, long long CoutStride, int ldc,
               int Ktot, int Ksplit,
               const float* __restrict__ embed, const int* __restrict__ txt,
               int xrow0, int xrow1,
               const unsigned short* __restrict__ hbf, long long hStride)
{
    const int dir = blockIdx.z;
    const unsigned short* Bm = Bmat + (long long)dir * BmatStride;
    const float* bias        = dir ? bias1 : bias0;
    float* C                 = Cout + (long long)dir * CoutStride;
    const int xrow           = dir ? xrow1 : xrow0;
    const unsigned short* hb = hbf + (long long)dir * hStride;

    __shared__ alignas(16) unsigned short As[128][40];   // [row][k], 16B row pad
    __shared__ alignas(16) unsigned short Bs[128][40];   // [col][k], 16B row pad

    const int tid  = threadIdx.x;
    const int lane = tid & 31;
    const int wave = tid >> 5;
    const int wm   = (wave >> 2) * 64;   // 0 / 64
    const int wn   = (wave & 3) * 32;    // 0 / 32 / 64 / 96
    const int rowBase = blockIdx.y * 128;
    const int colBase = blockIdx.x * 128;

    // cooperative LDS-fill assignment: 2 threads per row, 16 bf16 each
    const int arow  = tid >> 1;          // 0..127
    const int ahalf = (tid & 1) * 16;    // ushort offset 0 / 16
    const int gb    = rowBase + arow;    // global batch row for A

    int word = 0;
    if (Ksplit > 0) word = txt[(long long)xrow * BB + gb];

#if USE_TDM
    const unsigned asOff = (unsigned)(unsigned long long)&As[0][0];  // LDS byte offset
    const unsigned bsOff = (unsigned)(unsigned long long)&Bs[0][0];
#endif

    floatx8 acc[4][2];
    const floatx8 zero8 = {0.f,0.f,0.f,0.f,0.f,0.f,0.f,0.f};
    #pragma unroll
    for (int mi = 0; mi < 4; ++mi)
        #pragma unroll
        for (int ni = 0; ni < 2; ++ni) acc[mi][ni] = zero8;

    int k0 = 0;
    // ---------------- Phase 1: A = embedding gather ----------------
    for (; k0 < Ksplit; k0 += 32) {
        __syncthreads();
#if USE_TDM
        if (tid < 32)
            tdm_load_tile(bsOff, Bm + (long long)colBase * ldb + k0, (unsigned)ldb);
#else
        {
            const uint32x4* wsrc = (const uint32x4*)(Bm + (long long)(colBase + arow) * ldb + k0 + ahalf);
            *(uint32x4*)&Bs[arow][ahalf]     = wsrc[0];
            *(uint32x4*)&Bs[arow][ahalf + 8] = wsrc[1];
        }
#endif
        {   // gather one embed row slice, convert f32 -> bf16, stage to LDS
            const floatx4* fs = (const floatx4*)(embed + (long long)word * EE + k0 + ahalf);
            floatx4 f0 = fs[0], f1 = fs[1], f2 = fs[2], f3 = fs[3];
            uint32x4 d0, d1;
            d0.x = pack_bf2(f0.x, f0.y); d0.y = pack_bf2(f0.z, f0.w);
            d0.z = pack_bf2(f1.x, f1.y); d0.w = pack_bf2(f1.z, f1.w);
            d1.x = pack_bf2(f2.x, f2.y); d1.y = pack_bf2(f2.z, f2.w);
            d1.z = pack_bf2(f3.x, f3.y); d1.w = pack_bf2(f3.z, f3.w);
            *(uint32x4*)&As[arow][ahalf]     = d0;
            *(uint32x4*)&As[arow][ahalf + 8] = d1;
        }
#if USE_TDM
        if (tid < 32) wait_tensor0();
#endif
        __syncthreads();
        mma_tiles(As, Bs, lane, wm, wn, acc);
    }
    // ---------------- Phase 2: A = bf16 hidden state ----------------
    for (; k0 < Ktot; k0 += 32) {
        __syncthreads();
#if USE_TDM
        if (tid < 32) {
            tdm_load_tile(bsOff, Bm + (long long)colBase * ldb + k0, (unsigned)ldb);
            tdm_load_tile(asOff, hb + (long long)rowBase * WW + (k0 - Ksplit), (unsigned)WW);
            wait_tensor0();
        }
#else
        {
            const uint32x4* wsrc = (const uint32x4*)(Bm + (long long)(colBase + arow) * ldb + k0 + ahalf);
            *(uint32x4*)&Bs[arow][ahalf]     = wsrc[0];
            *(uint32x4*)&Bs[arow][ahalf + 8] = wsrc[1];
            const uint32x4* hs = (const uint32x4*)(hb + (long long)gb * WW + (k0 - Ksplit) + ahalf);
            *(uint32x4*)&As[arow][ahalf]     = hs[0];
            *(uint32x4*)&As[arow][ahalf + 8] = hs[1];
        }
#endif
        __syncthreads();
        mma_tiles(As, Bs, lane, wm, wn, acc);
    }

    // ---- epilogue: C/D layout -> lane = n + 16*(m/8), vgpr i = m%8 ----
    const int mlocal = (lane >> 4) << 3;   // 0 or 8
    const int ncol   = lane & 15;
    #pragma unroll
    for (int ni = 0; ni < 2; ++ni) {
        const int n = colBase + wn + ni * 16 + ncol;
        const float bv = bias[n];
        #pragma unroll
        for (int mi = 0; mi < 4; ++mi) {
            const int mg = rowBase + wm + mi * 16 + mlocal;
            float* cp = C + (long long)mg * ldc + n;
            #pragma unroll
            for (int i = 0; i < 8; ++i)
                cp[(long long)i * ldc] = acc[mi][ni][i] + bv;
        }
    }
}

// ---------------------------------------------------------------------------
// Pointwise GRU update (both directions via blockIdx.y).
// ---------------------------------------------------------------------------
__global__ __launch_bounds__(256)
void gru_pointwise(const float* __restrict__ G, const float* __restrict__ Ghn,
                   float* __restrict__ hstate, unsigned short* __restrict__ hbf,
                   const int* __restrict__ txt_len, int t, float* __restrict__ out_h)
{
    const int idx = blockIdx.x * 256 + threadIdx.x;   // 0 .. B*W-1
    const int dir = blockIdx.y;
    const int w = idx & (WW - 1);
    const int b = idx >> 10;

    const float* Gd  = G   + (long long)dir * BB * (3 * WW);
    const float* Hd  = Ghn + (long long)dir * BB * WW;
    float* hs        = hstate + (long long)dir * BB * WW;
    unsigned short* hbd = hbf + (long long)dir * BB * WW;

    const float gr  = Gd[(long long)b * (3 * WW) + w];
    const float gz  = Gd[(long long)b * (3 * WW) + WW + w];
    const float gn  = Gd[(long long)b * (3 * WW) + 2 * WW + w];
    const float ghn = Hd[idx];
    const float hp  = hs[idx];

    const float r = 1.f / (1.f + __expf(-gr));
    const float z = 1.f / (1.f + __expf(-gz));
    const float n = tanhf(gn + (r - 1.f) * ghn);
    const float hc = (1.f - z) * n + z * hp;

    const int len = txt_len[b];
    float hn;
    if (dir == 0) {
        hn = hc;                                      // forward: no mask on state
        const float m = (t < len) ? 1.f : 0.f;        // mask only on output
        out_h[(long long)t * (BB * WW) + idx] += 0.5f * hn * m;
    } else {
        const float m = ((TT - 1 - t) < len) ? 1.f : 0.f;
        hn = m * hc + (1.f - m) * hp;                 // masked hold
        out_h[(long long)(TT - 1 - t) * (BB * WW) + idx] += 0.5f * hn;
    }
    hs[idx]  = hn;
    hbd[idx] = f2bf(hn);
}

// ---------------------------------------------------------------------------
// Prep / utility kernels
// ---------------------------------------------------------------------------
__global__ void prep_wcat(const float* __restrict__ Wi_f, const float* __restrict__ Wh_f,
                          const float* __restrict__ Wi_b, const float* __restrict__ Wh_b,
                          unsigned short* __restrict__ Wcat)
{
    const long long idx = (long long)blockIdx.x * 256 + threadIdx.x;
    if (idx >= 2LL * (3 * WW) * (EE + WW)) return;
    const int k = (int)(idx & (EE + WW - 1));         // 0..2047
    const long long r = idx >> 11;
    const int n   = (int)(r % (3 * WW));
    const int dir = (int)(r / (3 * WW));
    const float* Wi = dir ? Wi_b : Wi_f;
    const float* Wh = dir ? Wh_b : Wh_f;
    const float v = (k < EE) ? Wi[(long long)n * EE + k]
                             : Wh[(long long)n * WW + (k - EE)];
    Wcat[idx] = f2bf(v);
}

__global__ void prep_bias(const float* __restrict__ bi_f, const float* __restrict__ bh_f,
                          const float* __restrict__ bi_b, const float* __restrict__ bh_b,
                          float* __restrict__ biasSum)
{
    const int idx = blockIdx.x * 256 + threadIdx.x;
    if (idx >= 2 * 3 * WW) return;
    const int n   = idx % (3 * WW);
    const int dir = idx / (3 * WW);
    const float* bi = dir ? bi_b : bi_f;
    const float* bh = dir ? bh_b : bh_f;
    biasSum[idx] = bi[n] + bh[n];
}

__global__ void zero_f32(float* __restrict__ p, long long n)
{
    const long long idx = (long long)blockIdx.x * 256 + threadIdx.x;
    if (idx < n) p[idx] = 0.f;
}

__global__ void mask_kernel(const int* __restrict__ txt_len, float* __restrict__ out_mask)
{
    const int idx = blockIdx.x * 256 + threadIdx.x;   // T*B
    if (idx >= TT * BB) return;
    const int t = idx >> 9;
    const int b = idx & (BB - 1);
    out_mask[idx] = (t < txt_len[b]) ? 1.f : 0.f;
}

__global__ void mout_kernel(const float* __restrict__ out_h,
                            const int* __restrict__ txt_len,
                            float* __restrict__ out_m)
{
    const int idx = blockIdx.x * 256 + threadIdx.x;   // B*W
    if (idx >= BB * WW) return;
    const int b = idx >> 10;
    float s = 0.f;
    #pragma unroll 4
    for (int t = 0; t < TT; ++t)
        s += out_h[(long long)t * (BB * WW) + idx];
    out_m[idx] = s / (float)txt_len[b];
}

// ---------------------------------------------------------------------------
// Workspace layout (bytes, 256-aligned)
// ---------------------------------------------------------------------------
static constexpr long long OFF_WCAT = 0;                              // 2*3072*2048 u16 = 25165824
static constexpr long long OFF_BIAS = 25165824;                       // 2*3072 f32
static constexpr long long OFF_G    = 25190400;                       // 2*512*3072 f32 = 12582912
static constexpr long long OFF_GHN  = 37773312;                       // 2*512*1024 f32 = 4194304
static constexpr long long OFF_HF   = 41967616;                       // 2*512*1024 f32 = 4194304
static constexpr long long OFF_HB   = 46161920;                       // 2*512*1024 u16 = 2097152
// total ~48.3 MB

extern "C" void kernel_launch(void* const* d_in, const int* in_sizes, int n_in,
                              void* d_out, int out_size, void* d_ws, size_t ws_size,
                              hipStream_t stream)
{
    const int*   txt     = (const int*)d_in[0];
    const int*   txt_len = (const int*)d_in[1];
    const float* embed   = (const float*)d_in[2];
    const float* Wi_f    = (const float*)d_in[3];
    const float* Wh_f    = (const float*)d_in[4];
    const float* bi_f    = (const float*)d_in[5];
    const float* bh_f    = (const float*)d_in[6];
    const float* Wi_b    = (const float*)d_in[7];
    const float* Wh_b    = (const float*)d_in[8];
    const float* bi_b    = (const float*)d_in[9];
    const float* bh_b    = (const float*)d_in[10];
    float* out = (float*)d_out;

    char* ws = (char*)d_ws;
    unsigned short* Wcat   = (unsigned short*)(ws + OFF_WCAT);
    float*          biasSum= (float*)(ws + OFF_BIAS);
    float*          G      = (float*)(ws + OFF_G);
    float*          Ghn    = (float*)(ws + OFF_GHN);
    float*          hF     = (float*)(ws + OFF_HF);
    unsigned short* hB     = (unsigned short*)(ws + OFF_HB);

    float* out_h    = out;                                  // T*B*W
    float* out_m    = out + (long long)TT * BB * WW;        // B*W
    float* out_mask = out_m + (long long)BB * WW;           // T*B

    // ---- prep ----
    {
        long long n = 2LL * (3 * WW) * (EE + WW);
        prep_wcat<<<dim3((unsigned)((n + 255) / 256)), 256, 0, stream>>>(Wi_f, Wh_f, Wi_b, Wh_b, Wcat);
    }
    prep_bias<<<dim3((2 * 3 * WW + 255) / 256), 256, 0, stream>>>(bi_f, bh_f, bi_b, bh_b, biasSum);
    {
        long long n = 2LL * BB * WW;
        zero_f32<<<dim3((unsigned)((n + 255) / 256)), 256, 0, stream>>>(hF, n);
        zero_f32<<<dim3((unsigned)((n / 2 + 255) / 256)), 256, 0, stream>>>((float*)hB, n / 2);
    }
    {
        long long n = (long long)TT * BB * WW;
        zero_f32<<<dim3((unsigned)((n + 255) / 256)), 256, 0, stream>>>(out_h, n);
    }
    mask_kernel<<<dim3((TT * BB + 255) / 256), 256, 0, stream>>>(txt_len, out_mask);

    // ---- sequential recurrence ----
    for (int t = 0; t < TT; ++t) {
        // gates GEMM: C = [x_t | h] @ [Wi ; Wh].T + (bi+bh), N=3072, K=2048
        gemm_step<<<dim3(24, 4, 2), 256, 0, stream>>>(
            Wcat, 3LL * WW * (EE + WW), EE + WW,
            biasSum, biasSum + 3 * WW,
            G, (long long)BB * 3 * WW, 3 * WW,
            EE + WW, EE,
            embed, txt, t, TT - 1 - t,
            hB, (long long)BB * WW);
        // Ghn GEMM: h @ Wh_n.T + bh_n, N=1024, K=1024 (Wcat rows 2048.., cols 1024..)
        gemm_step<<<dim3(8, 4, 2), 256, 0, stream>>>(
            Wcat + (long long)(2 * WW) * (EE + WW) + EE, 3LL * WW * (EE + WW), EE + WW,
            bh_f + 2 * WW, bh_b + 2 * WW,
            Ghn, (long long)BB * WW, WW,
            WW, 0,
            (const float*)nullptr, (const int*)nullptr, 0, 0,
            hB, (long long)BB * WW);
        gru_pointwise<<<dim3((BB * WW) / 256, 2), 256, 0, stream>>>(
            G, Ghn, hF, hB, txt_len, t, out_h);
    }

    // ---- m_out reduction ----
    mout_kernel<<<dim3((BB * WW + 255) / 256), 256, 0, stream>>>(out_h, txt_len, out_m);
}
